// ReasoningModule_43714177139357
// MI455X (gfx1250) — 8-bit matrix operand order repaired and checked by reading, compiled, not run on hardware
//
#include <hip/hip_runtime.h>
#include <math.h>

typedef __attribute__((ext_vector_type(8))) int   v8i;
typedef __attribute__((ext_vector_type(8))) float v8f;
typedef __attribute__((ext_vector_type(2))) float v2f;
typedef __attribute__((ext_vector_type(4))) float v4f;

#define B_  32
#define S_  2048
#define E_  1024
#define R_  512
#define STEPS_ 8

#define XM_OFF   0
#define ST_OFF   32768
#define CUR_OFF  32784
#define H0_OFF   49168
#define C0_OFF   65552
#define H1_OFF   81936
#define C1_OFF   98320
#define ACC_OFF  114704
#define ROUT_OFF 131088
#define TOT_FLOATS 163856
#define ACTIVE_IDX (ST_OFF + 4)

__device__ __forceinline__ float sigmoidf_(float x) { return 1.f / (1.f + __expf(-x)); }

__device__ __forceinline__ int pack4_tern(const float* w, float thresh) {
  int r = 0;
#pragma unroll
  for (int j = 0; j < 4; ++j) {
    float v = w[j];
    int q = (fabsf(v) > thresh) ? (v > 0.f ? 1 : -1) : 0;
    r |= (q & 0xFF) << (8 * j);
  }
  return r;
}

__device__ __forceinline__ v8i load_a_i8(const signed char* Q, int ldk, int m0, int k0, int lane) {
  int row = m0 + (lane & 15);
  int half8 = (lane >> 4) << 3;
  const signed char* base = Q + row * ldk + k0 + half8;
  v8i a;
#pragma unroll
  for (int j = 0; j < 8; ++j) {
    int koff = ((j >> 1) << 4) + ((j & 1) << 2);
    a[j] = *(const int*)(base + koff);
  }
  return a;
}

__device__ __forceinline__ v8i load_bT_i8q(const float* W, int ldw, int k0, int n0, int lane,
                                           float thresh) {
  int n = n0 + (lane & 15);
  int khalf = (lane >> 4) << 3;
  const float* base = W + (size_t)n * ldw + k0 + khalf;
  v8i b;
#pragma unroll
  for (int j = 0; j < 8; ++j) {
    int koff = ((j >> 1) << 4) + ((j & 1) << 2);
    b[j] = pack4_tern(base + koff, thresh);
  }
  return b;
}

__device__ __forceinline__ v2f load_a_f32(const float* A, int lda, int m0, int k0, int lane) {
  int row = m0 + (lane & 15);
  int kk = k0 + ((lane >> 4) << 1);
  const float* p = A + (size_t)row * lda + kk;
  v2f a; a.x = p[0]; a.y = p[1];
  return a;
}

__device__ __forceinline__ v2f load_bT_f32(const float* W, int ldw, int k0, int n0, int lane) {
  int n = n0 + (lane & 15);
  int kk = k0 + ((lane >> 4) << 1);
  const float* p = W + (size_t)n * ldw + kk;
  v2f b; b.x = p[0]; b.y = p[1];
  return b;
}

__device__ __forceinline__ v8f wmma_f32x4(v2f a, v2f b, v8f c) {
  return __builtin_amdgcn_wmma_f32_16x16x4_f32(false, a, false, b, (short)0, c, false, false);
}
__device__ __forceinline__ v8i wmma_iu8(v8i a, v8i b, v8i c) {
  return __builtin_amdgcn_wmma_i32_16x16x64_iu8(true, a, true, b, c, false, false);
}


__global__ void init_ws_kernel(float* ws) {
  int i = blockIdx.x * 256 + threadIdx.x;
  if (i < TOT_FLOATS) ws[i] = (i == ACTIVE_IDX) ? 1.f : 0.f;
}

__global__ void mean_partial_kernel(const float* __restrict__ x, float* __restrict__ xm) {
  int b = blockIdx.x;
  int tid = threadIdx.x;
  const v4f* p = (const v4f*)(x + ((size_t)b * S_ + (size_t)blockIdx.y * 128) * E_) + tid;
  v4f s = {0.f, 0.f, 0.f, 0.f};
#pragma unroll 4
  for (int i = 0; i < 128; ++i) {
    v4f v = __builtin_nontemporal_load(p + (size_t)i * (E_ / 4));
    s += v;
  }
  float* o = &xm[b * E_ + tid * 4];
  atomicAdd(o + 0, s.x * (1.f / (float)S_));
  atomicAdd(o + 1, s.y * (1.f / (float)S_));
  atomicAdd(o + 2, s.z * (1.f / (float)S_));
  atomicAdd(o + 3, s.w * (1.f / (float)S_));
}

__global__ void absmax_div127_kernel(const float* __restrict__ v, int n, float* out) {
  __shared__ float red[256];
  float m = 0.f;
  for (int i = threadIdx.x; i < n; i += 256) m = fmaxf(m, fabsf(v[i]));
  red[threadIdx.x] = m; __syncthreads();
  for (int s = 128; s > 0; s >>= 1) {
    if (threadIdx.x < s) red[threadIdx.x] = fmaxf(red[threadIdx.x], red[threadIdx.x + s]);
    __syncthreads();
  }
  if (threadIdx.x == 0) out[0] = red[0] * (1.f / 127.f);
}

__global__ void wscale_kernel(const float* encW, const float* decW, const float* gateW,
                              float* st) {
  const float* W; int n; int slot;
  if (blockIdx.x == 0)      { W = encW;  n = R_ * E_; slot = 1; }
  else if (blockIdx.x == 1) { W = decW;  n = E_ * R_; slot = 2; }
  else                      { W = gateW; n = R_;      slot = 3; }
  __shared__ float red[256];
  float s = 0.f;
  for (int i = threadIdx.x; i < n; i += 256) s += fabsf(W[i]);
  red[threadIdx.x] = s; __syncthreads();
  for (int k = 128; k > 0; k >>= 1) {
    if (threadIdx.x < k) red[threadIdx.x] += red[threadIdx.x + k];
    __syncthreads();
  }
  if (threadIdx.x == 0) st[slot] = red[0] / (float)n;
}

__global__ void encoder_kernel(const float* __restrict__ xm, const float* __restrict__ encW,
                               const float* __restrict__ encb, const float* __restrict__ st,
                               float* __restrict__ cur) {
  __shared__ signed char q[B_ * E_];
  float iscale = st[0];
  float inv = 1.f / iscale;
  float thresh = 0.5f * st[1];
  float scale = st[1] * iscale;
  for (int i = threadIdx.x; i < B_ * E_; i += 128) {
    float v = fminf(127.f, fmaxf(-128.f, rintf(xm[i] * inv)));
    q[i] = (signed char)(int)v;
  }
  __syncthreads();
  int lane = threadIdx.x & 31, wave = threadIdx.x >> 5;
  int t = blockIdx.x * 4 + wave;
  int m0 = (t & 1) << 4, n0 = (t >> 1) << 4;
  v8i acc = {};
  for (int k0 = 0; k0 < E_; k0 += 64) {
    v8i a = load_a_i8(q, E_, m0, k0, lane);
    v8i b = load_bT_i8q(encW, E_, k0, n0, lane, thresh);
    acc = wmma_iu8(a, b, acc);
  }
  int n = n0 + (lane & 15);
  int rb = m0 + ((lane >> 4) << 3);
#pragma unroll
  for (int r = 0; r < 8; ++r)
    cur[(rb + r) * R_ + n] = (float)acc[r] * scale + encb[n];
}

__global__ void lstm_step_kernel(const float* __restrict__ cur, const float* __restrict__ hin,
                                 const float* __restrict__ cin, const float* __restrict__ Wih,
                                 const float* __restrict__ Whh, const float* __restrict__ bih,
                                 const float* __restrict__ bhh, float* __restrict__ hout,
                                 float* __restrict__ cout) {
  int lane = threadIdx.x & 31, wave = threadIdx.x >> 5;
  int t = blockIdx.x * 4 + wave;
  int m0 = (t & 1) << 4, n0 = (t >> 1) << 4;
  v8f acc[4] = {v8f{}, v8f{}, v8f{}, v8f{}};
  for (int k0 = 0; k0 < R_; k0 += 4) {
    v2f ac = load_a_f32(cur, R_, m0, k0, lane);
    v2f ah = load_a_f32(hin, R_, m0, k0, lane);
#pragma unroll
    for (int p = 0; p < 4; ++p) {
      v2f b1 = load_bT_f32(Wih, R_, k0, p * R_ + n0, lane);
      acc[p] = wmma_f32x4(ac, b1, acc[p]);
      v2f b2 = load_bT_f32(Whh, R_, k0, p * R_ + n0, lane);
      acc[p] = wmma_f32x4(ah, b2, acc[p]);
    }
  }
  int n = n0 + (lane & 15);
  int rb = m0 + ((lane >> 4) << 3);
  float bs0 = bih[0 * R_ + n] + bhh[0 * R_ + n];
  float bs1 = bih[1 * R_ + n] + bhh[1 * R_ + n];
  float bs2 = bih[2 * R_ + n] + bhh[2 * R_ + n];
  float bs3 = bih[3 * R_ + n] + bhh[3 * R_ + n];
#pragma unroll
  for (int r = 0; r < 8; ++r) {
    float iv = sigmoidf_(acc[0][r] + bs0);
    float fv = sigmoidf_(acc[1][r] + bs1);
    float gv = tanhf(acc[2][r] + bs2);
    float ov = sigmoidf_(acc[3][r] + bs3);
    int idx = (rb + r) * R_ + n;
    float cn = fv * cin[idx] + iv * gv;
    float hn = ov * tanhf(cn);
    cout[idx] = cn;
    hout[idx] = hn;
  }
}

__global__ void gate_acc_kernel(const float* __restrict__ h, const float* __restrict__ gateW,
                                const float* __restrict__ gateb, float* __restrict__ st,
                                float* __restrict__ acc, int step) {
  __shared__ float red[512];
  __shared__ float scores[B_];
  int tid = threadIdx.x;
  float active = st[4];
  float m = 0.f;
  for (int i = tid; i < B_ * R_; i += 512) m = fmaxf(m, fabsf(h[i]));
  red[tid] = m; __syncthreads();
  for (int s = 256; s > 0; s >>= 1) {
    if (tid < s) red[tid] = fmaxf(red[tid], red[tid + s]);
    __syncthreads();
  }
  float iscale = red[0] * (1.f / 127.f);
  float inv = 1.f / iscale;
  __syncthreads();
  for (int i = tid; i < B_ * R_; i += 512) acc[i] += active * h[i];
  float thresh = 0.5f * st[3];
  int b = tid >> 4;
  int c0 = (tid & 15) << 5;
  float s = 0.f;
  for (int r = c0; r < c0 + 32; ++r) {
    float qx = fminf(127.f, fmaxf(-128.f, rintf(h[b * R_ + r] * inv)));
    float w = gateW[r];
    float qw = (fabsf(w) > thresh) ? (w > 0.f ? 1.f : -1.f) : 0.f;
    s += qx * qw;
  }
  red[tid] = s; __syncthreads();
  if ((tid & 15) == 0) {
    float tot = 0.f;
#pragma unroll
    for (int j = 0; j < 16; ++j) tot += red[tid + j];
    scores[b] = sigmoidf_(tot * st[3] * iscale + gateb[0]);
  }
  __syncthreads();
  if (tid == 0) {
    bool all_lt = true;
    for (int bb = 0; bb < B_; ++bb) all_lt = all_lt && (scores[bb] < 0.5f);
    st[5] += active;
    if (all_lt && (step > 2)) st[4] = 0.f;
  }
}

__global__ void finalize_kernel(const float* __restrict__ acc, float* __restrict__ st,
                                float* __restrict__ fin) {
  __shared__ float red[256];
  float inv = 1.f / st[5];
  float m = 0.f;
  for (int i = threadIdx.x; i < B_ * R_; i += 256) {
    float v = acc[i] * inv;
    fin[i] = v;
    m = fmaxf(m, fabsf(v));
  }
  red[threadIdx.x] = m; __syncthreads();
  for (int s = 128; s > 0; s >>= 1) {
    if (threadIdx.x < s) red[threadIdx.x] = fmaxf(red[threadIdx.x], red[threadIdx.x + s]);
    __syncthreads();
  }
  if (threadIdx.x == 0) st[6] = red[0] * (1.f / 127.f);
}

__global__ void decoder_kernel(const float* __restrict__ fin, const float* __restrict__ decW,
                               const float* __restrict__ decb, const float* __restrict__ st,
                               float* __restrict__ rout) {
  __shared__ signed char q[B_ * R_];
  float iscale = st[6];
  float inv = 1.f / iscale;
  float thresh = 0.5f * st[2];
  float scale = st[2] * iscale;
  for (int i = threadIdx.x; i < B_ * R_; i += 128) {
    float v = fminf(127.f, fmaxf(-128.f, rintf(fin[i] * inv)));
    q[i] = (signed char)(int)v;
  }
  __syncthreads();
  int lane = threadIdx.x & 31, wave = threadIdx.x >> 5;
  int t = blockIdx.x * 4 + wave;
  int m0 = (t & 1) << 4, n0 = (t >> 1) << 4;
  v8i acc = {};
  for (int k0 = 0; k0 < R_; k0 += 64) {
    v8i a = load_a_i8(q, R_, m0, k0, lane);
    v8i b = load_bT_i8q(decW, R_, k0, n0, lane, thresh);
    acc = wmma_iu8(a, b, acc);
  }
  int n = n0 + (lane & 15);
  int rb = m0 + ((lane >> 4) << 3);
#pragma unroll
  for (int r = 0; r < 8; ++r)
    rout[(rb + r) * E_ + n] = (float)acc[r] * scale + decb[n];
}

__global__ void bcast_add_kernel(const v4f* __restrict__ x4, const float* __restrict__ rout,
                                 v4f* __restrict__ o4) {
  int idx = blockIdx.x * 256 + threadIdx.x;
  int b = idx >> 19;
  v4f v = __builtin_nontemporal_load(x4 + idx);
  const v4f* r4 = (const v4f*)rout;
  v4f rv = r4[(b << 8) + (idx & 255)];
  v += rv;
  __builtin_nontemporal_store(v, o4 + idx);
}

extern "C" void kernel_launch(void* const* d_in, const int* in_sizes, int n_in,
                              void* d_out, int out_size, void* d_ws, size_t ws_size,
                              hipStream_t stream) {
  (void)in_sizes; (void)n_in; (void)out_size; (void)ws_size;
  const float* x     = (const float*)d_in[0];
  const float* encW  = (const float*)d_in[1];
  const float* encb  = (const float*)d_in[2];
  const float* Wih   = (const float*)d_in[3];
  const float* Whh   = (const float*)d_in[4];
  const float* bih   = (const float*)d_in[5];
  const float* bhh   = (const float*)d_in[6];
  const float* decW  = (const float*)d_in[7];
  const float* decb  = (const float*)d_in[8];
  const float* gateW = (const float*)d_in[9];
  const float* gateb = (const float*)d_in[10];
  float* out = (float*)d_out;
  float* ws  = (float*)d_ws;

  init_ws_kernel<<<(TOT_FLOATS + 255) / 256, 256, 0, stream>>>(ws);
  mean_partial_kernel<<<dim3(B_, S_ / 128), 256, 0, stream>>>(x, ws + XM_OFF);
  absmax_div127_kernel<<<1, 256, 0, stream>>>(ws + XM_OFF, B_ * E_, ws + ST_OFF + 0);
  wscale_kernel<<<3, 256, 0, stream>>>(encW, decW, gateW, ws + ST_OFF);
  encoder_kernel<<<16, 128, 0, stream>>>(ws + XM_OFF, encW, encb, ws + ST_OFF, ws + CUR_OFF);

  for (int t = 0; t < STEPS_; ++t) {
    float* hin  = ws + ((t & 1) ? H1_OFF : H0_OFF);
    float* cin  = ws + ((t & 1) ? C1_OFF : C0_OFF);
    float* hout = ws + ((t & 1) ? H0_OFF : H1_OFF);
    float* cout = ws + ((t & 1) ? C0_OFF : C1_OFF);
    const float* cp = (t == 0) ? (ws + CUR_OFF) : hin;
    lstm_step_kernel<<<16, 128, 0, stream>>>(cp, hin, cin, Wih, Whh, bih, bhh, hout, cout);
    gate_acc_kernel<<<1, 512, 0, stream>>>(hout, gateW, gateb, ws + ST_OFF, ws + ACC_OFF, t);
  }

  finalize_kernel<<<1, 256, 0, stream>>>(ws + ACC_OFF, ws + ST_OFF, ws + CUR_OFF);
  decoder_kernel<<<32, 128, 0, stream>>>(ws + CUR_OFF, decW, decb, ws + ST_OFF, ws + ROUT_OFF);
  bcast_add_kernel<<<(B_ * S_ * E_ / 4) / 256, 256, 0, stream>>>(
      (const v4f*)x, ws + ROUT_OFF, (v4f*)out);
}
